// Anonymizer_39505109189281
// MI455X (gfx1250) — matrix operand order repaired and checked by reading, not compiled, not run on hardware
//
#include <hip/hip_runtime.h>
#include <hip/hip_bf16.h>

// ---------------------------------------------------------------------------
// Seq2seq LSTM (encoder 300 steps + autoregressive decoder 299 steps) for
// MI455X (gfx1250, wave32, WMMA). Latency-bound serial recurrence:
//  - one workgroup per 16-row batch tile (batch rows independent), entire
//    encoder+decoder runs inside the persistent workgroup, h/c/feedback in LDS
//  - weights f16, LDS-resident, pre-swizzled into WMMA B-fragment order
//    (2x ds_read_b128 per fragment), gates via v_wmma_f32_16x16x32_f16
//  - per-step critical path minimized: two independent WMMA accumulate chains
//    (Wih: 5 deep, Whh: 4 deep) merged with one VALU add; encoder x-tile is
//    register-staged/double-buffered so HBM latency overlaps compute
// Dims: N=512 T=300 F=150 H=100 G=400; pads F->160 (5 ksteps), H->128 (4).
// 25 gate tiles -> 25 waves (800 threads), one 16x16 tile per wave.
// LDS: 299.5 KB < 320 KB per WGP.
// ---------------------------------------------------------------------------

typedef __attribute__((ext_vector_type(16))) _Float16 v16h;
typedef __attribute__((ext_vector_type(8)))  _Float16 v8h;
typedef __attribute__((ext_vector_type(8)))  float    v8f;

#define NBATCH    512
#define TSTEPS    300
#define FDIM      150
#define HDIM      100
#define GDIM      400          // 4*H
#define FP        160          // F padded (5 * 32)
#define HP        128          // H padded (4 * 32)
#define KS_F      5
#define KS_H      4
#define NT_G      25           // gate tiles (400/16)
#define NT_FC     10           // fc output tiles (160/16)
#define NTHREADS  800          // 25 waves of 32
#define XLOADS    3            // 16*150 / 800
#define XB_N      45000        // C*T*V*M (batch stride in x/out)
#define XB_C      15000        // T*V*M   (channel stride)

// LDS layout (bytes)
#define OFF_WIH   0                         // 25*5*32*16*2 = 128000
#define OFF_WHH   128000                    // 25*4*32*16*2 = 102400
#define OFF_FC    230400                    // 10*4*32*16*2 = 40960
#define OFF_GATES 250880                    // 16*400*4     = 25600
#define OFF_C     276480                    // 16*100*4     = 6400
#define OFF_H     282880                    // 16*128*2     = 4096
#define OFF_IN0   286976                    // 16*160*2     = 5120
#define OFF_IN1   292096                    // 16*160*2     = 5120
#define OFF_BSUM  297216                    // 400*4        = 1600
#define OFF_FCB   298816                    // 160*4        = 640
#define SMEM_BYTES 299456

__device__ __forceinline__ float sigmoidf_fast(float v) {
    return 1.0f / (1.0f + __expf(-v));
}

// A-fragment (16x32 f16) from row-major LDS buffer [16][pitch]:
// lanes 0-15: row=lane,   halves K = kb+0..7 and kb+16..23 (kb = ks*32)
// lanes16-31: row=lane-16, kb offset +8
__device__ __forceinline__ v16h load_a(const _Float16* __restrict__ buf,
                                       int pitch, int ks, int lane) {
    int row = lane & 15;
    int kb  = ks * 32 + ((lane >> 4) << 3);
    v8h lo = *(const v8h*)(buf + row * pitch + kb);
    v8h hi = *(const v8h*)(buf + row * pitch + kb + 16);
    v16h a;
#pragma unroll
    for (int i = 0; i < 8; ++i) { a[i] = lo[i]; a[8 + i] = hi[i]; }
    return a;
}

// B-fragment (32x16 f16) from pre-swizzled LDS: contiguous 32B per lane.
__device__ __forceinline__ v16h load_b(const _Float16* __restrict__ wf,
                                       int ksteps, int nt, int ks, int lane) {
    const _Float16* q = wf + (((nt * ksteps + ks) * 32 + lane) << 4);
    v8h b0 = *(const v8h*)q;
    v8h b1 = *(const v8h*)(q + 8);
    v16h b;
#pragma unroll
    for (int i = 0; i < 8; ++i) { b[i] = b0[i]; b[8 + i] = b1[i]; }
    return b;
}

// C/D layout: VGPR v -> row v (lanes 0-15) / row v+8 (lanes 16-31), col=lane&15
__device__ __forceinline__ void store_acc2(float* __restrict__ dst, int pitch,
                                           int nt, int lane, v8f a0, v8f a1) {
    int col = nt * 16 + (lane & 15);
    int rb  = (lane >> 4) << 3;
#pragma unroll
    for (int v = 0; v < 8; ++v) dst[(rb + v) * pitch + col] = a0[v] + a1[v];
}
__device__ __forceinline__ void store_acc(float* __restrict__ dst, int pitch,
                                          int nt, int lane, v8f a0) {
    int col = nt * 16 + (lane & 15);
    int rb  = (lane >> 4) << 3;
#pragma unroll
    for (int v = 0; v < 8; ++v) dst[(rb + v) * pitch + col] = a0[v];
}

// Fill a B-fragment weight region from global row-major W[ncols][kdim]:
// value at (nt,ks,lane,j) = W[g][K], g = nt*16+(lane&15),
// K = ks*32 + ((lane>=16)?8:0) + (j&7) + ((j>=8)?16:0) ; zero outside kdim/ncols.
__device__ __forceinline__ void fill_b(_Float16* __restrict__ dst,
                                       const float* __restrict__ src,
                                       int ntiles, int ksteps, int kdim,
                                       int ncols, int tid) {
    int total = ntiles * ksteps * 32 * 16;
    for (int idx = tid; idx < total; idx += NTHREADS) {
        int j  = idx & 15;
        int l  = (idx >> 4) & 31;
        int tk = idx >> 9;               // nt*ksteps + ks
        int ks = tk % ksteps;
        int nt = tk / ksteps;
        int K  = ks * 32 + ((l >> 4) << 3) + (j & 7) + ((j >> 3) << 4);
        int g  = nt * 16 + (l & 15);
        float v = (K < kdim && g < ncols) ? src[g * kdim + K] : 0.0f;
        dst[idx] = (_Float16)v;
    }
}

__global__ void __launch_bounds__(NTHREADS, 1)
seq2seq_lstm_kernel(const float* __restrict__ x,
                    const float* __restrict__ encWih, const float* __restrict__ encWhh,
                    const float* __restrict__ encBih, const float* __restrict__ encBhh,
                    const float* __restrict__ decWih, const float* __restrict__ decWhh,
                    const float* __restrict__ decBih, const float* __restrict__ decBhh,
                    const float* __restrict__ fcW,    const float* __restrict__ fcB,
                    float* __restrict__ out) {
    extern __shared__ char smem[];
    _Float16* sWih  = (_Float16*)(smem + OFF_WIH);
    _Float16* sWhh  = (_Float16*)(smem + OFF_WHH);
    _Float16* sFc   = (_Float16*)(smem + OFF_FC);
    float*    sGate = (float*)(smem + OFF_GATES);   // [16][400]
    float*    sC    = (float*)(smem + OFF_C);       // [16][100]
    _Float16* sH    = (_Float16*)(smem + OFF_H);    // [16][128], padded zeros
    _Float16* sIn0  = (_Float16*)(smem + OFF_IN0);  // [16][160], ping
    _Float16* sIn1  = (_Float16*)(smem + OFF_IN1);  // [16][160], pong
    float*    sBsum = (float*)(smem + OFF_BSUM);    // [400]
    float*    sFcb  = (float*)(smem + OFF_FCB);     // [160]

    const int tid  = threadIdx.x;
    const int lane = tid & 31;
    const int wid  = tid >> 5;          // 0..24
    const int n0   = blockIdx.x * 16;   // batch tile base

    // per-thread x gather coordinates (e = tid + q*NTHREADS, e = r*150 + f)
    int xr_[XLOADS], xo_[XLOADS];       // row, (c*XB_C + w) offsets
#pragma unroll
    for (int q = 0; q < XLOADS; ++q) {
        int e = tid + q * NTHREADS;
        int r = e / FDIM, f = e % FDIM;
        xr_[q] = r;
        xo_[q] = (f / 50) * XB_C + (f % 50);
    }
    int inoff_[XLOADS];
#pragma unroll
    for (int q = 0; q < XLOADS; ++q) {
        int e = tid + q * NTHREADS;
        inoff_[q] = (e / FDIM) * FP + (e % FDIM);
    }

    // ---- init: states, padding, encoder bias-sum, t=0 output slice ----
    for (int i = tid; i < 16 * HP;  i += NTHREADS) sH[i]   = (_Float16)0.0f;
    for (int i = tid; i < 16 * FP;  i += NTHREADS) sIn0[i] = (_Float16)0.0f;
    for (int i = tid; i < 16 * FP;  i += NTHREADS) sIn1[i] = (_Float16)0.0f;
    for (int i = tid; i < 16 * HDIM; i += NTHREADS) sC[i]  = 0.0f;
    for (int i = tid; i < GDIM; i += NTHREADS) sBsum[i] = encBih[i] + encBhh[i];
    for (int e = tid; e < 16 * FDIM; e += NTHREADS) {
        int r = e / FDIM, f = e % FDIM;
        out[(size_t)(n0 + r) * XB_N + (f / 50) * XB_C + (f % 50)] = 0.0f;
    }
    // encoder weights -> LDS fragments (f16)
    fill_b(sWih, encWih, NT_G, KS_F, FDIM, GDIM, tid);
    fill_b(sWhh, encWhh, NT_G, KS_H, HDIM, GDIM, tid);

    // preload x_0 into ping buffer
#pragma unroll
    for (int q = 0; q < XLOADS; ++q)
        sIn0[inoff_[q]] =
            (_Float16)x[(size_t)(n0 + xr_[q]) * XB_N + xo_[q] /* t=0 */];
    __syncthreads();

    // ======================= ENCODER: t = 0..T-1 ==========================
    for (int t = 0; t < TSTEPS; ++t) {
        const _Float16* sInCur = (t & 1) ? sIn1 : sIn0;
        _Float16*       sInNxt = (t & 1) ? sIn0 : sIn1;

        // issue next step's x gathers now; waits overlap this step's compute
        float xreg[XLOADS];
        if (t + 1 < TSTEPS) {
#pragma unroll
            for (int q = 0; q < XLOADS; ++q)
                xreg[q] = x[(size_t)(n0 + xr_[q]) * XB_N + (t + 1) * 50 + xo_[q]];
            // warm L2 for t+2 (WGP-scope prefetch)
            if (t + 2 < TSTEPS)
                __builtin_prefetch(
                    &x[(size_t)(n0 + xr_[0]) * XB_N + (t + 2) * 50 + xo_[0]], 0, 3);
        }

        // gates[16x400] = x_t @ Wih^T + h @ Whh^T, two independent WMMA chains
        {
            v8f acc0 = {}, acc1 = {};
#pragma unroll
            for (int ks = 0; ks < KS_F; ++ks) {
                v16h a = load_a(sInCur, FP, ks, lane);
                v16h b = load_b(sWih, KS_F, wid, ks, lane);
                acc0 = __builtin_amdgcn_wmma_f32_16x16x32_f16(false, a, false, b,
                                                              (short)0, acc0, false, false);
            }
#pragma unroll
            for (int ks = 0; ks < KS_H; ++ks) {
                v16h a = load_a(sH, HP, ks, lane);
                v16h b = load_b(sWhh, KS_H, wid, ks, lane);
                acc1 = __builtin_amdgcn_wmma_f32_16x16x32_f16(false, a, false, b,
                                                              (short)0, acc1, false, false);
            }
            store_acc2(sGate, GDIM, wid, lane, acc0, acc1);
        }
        __syncthreads();

        // elementwise LSTM cell update (i,f,g,o), h -> f16 A-buffer
        for (int e = tid; e < 16 * HDIM; e += NTHREADS) {
            int r = e / HDIM, j = e % HDIM;
            const float* gr = sGate + r * GDIM;
            float iv = sigmoidf_fast(gr[j]            + sBsum[j]);
            float fv = sigmoidf_fast(gr[HDIM + j]     + sBsum[HDIM + j]);
            float gv = tanhf(        gr[2 * HDIM + j] + sBsum[2 * HDIM + j]);
            float ov = sigmoidf_fast(gr[3 * HDIM + j] + sBsum[3 * HDIM + j]);
            float cn = fv * sC[r * HDIM + j] + iv * gv;
            sC[r * HDIM + j] = cn;
            sH[r * HP + j] = (_Float16)(ov * tanhf(cn));
        }
        // drain register-staged x_{t+1} into the pong buffer
        if (t + 1 < TSTEPS) {
#pragma unroll
            for (int q = 0; q < XLOADS; ++q)
                sInNxt[inoff_[q]] = (_Float16)xreg[q];
        }
        __syncthreads();
    }

    // ---- swap in decoder weights over the same LDS region ----
    fill_b(sWih, decWih, NT_G, KS_F, FDIM, GDIM, tid);
    fill_b(sWhh, decWhh, NT_G, KS_H, HDIM, GDIM, tid);
    fill_b(sFc,  fcW,    NT_FC, KS_H, HDIM, FDIM, tid);
    for (int i = tid; i < GDIM; i += NTHREADS) sBsum[i] = decBih[i] + decBhh[i];
    for (int i = tid; i < FP; i += NTHREADS) sFcb[i] = (i < FDIM) ? fcB[i] : 0.0f;
    for (int i = tid; i < 16 * FP; i += NTHREADS) sIn0[i] = (_Float16)0.0f; // inp0=0
    __syncthreads();

    // ================= DECODER (autoregressive): t = 1..T-1 ===============
    for (int t = 1; t < TSTEPS; ++t) {
        // gates = inp @ decWih^T + h @ decWhh^T (two independent chains)
        {
            v8f acc0 = {}, acc1 = {};
#pragma unroll
            for (int ks = 0; ks < KS_F; ++ks) {
                v16h a = load_a(sIn0, FP, ks, lane);
                v16h b = load_b(sWih, KS_F, wid, ks, lane);
                acc0 = __builtin_amdgcn_wmma_f32_16x16x32_f16(false, a, false, b,
                                                              (short)0, acc0, false, false);
            }
#pragma unroll
            for (int ks = 0; ks < KS_H; ++ks) {
                v16h a = load_a(sH, HP, ks, lane);
                v16h b = load_b(sWhh, KS_H, wid, ks, lane);
                acc1 = __builtin_amdgcn_wmma_f32_16x16x32_f16(false, a, false, b,
                                                              (short)0, acc1, false, false);
            }
            store_acc2(sGate, GDIM, wid, lane, acc0, acc1);
        }
        __syncthreads();

        // LSTM cell update
        for (int e = tid; e < 16 * HDIM; e += NTHREADS) {
            int r = e / HDIM, j = e % HDIM;
            const float* gr = sGate + r * GDIM;
            float iv = sigmoidf_fast(gr[j]            + sBsum[j]);
            float fv = sigmoidf_fast(gr[HDIM + j]     + sBsum[HDIM + j]);
            float gv = tanhf(        gr[2 * HDIM + j] + sBsum[2 * HDIM + j]);
            float ov = sigmoidf_fast(gr[3 * HDIM + j] + sBsum[3 * HDIM + j]);
            float cn = fv * sC[r * HDIM + j] + iv * gv;
            sC[r * HDIM + j] = cn;
            sH[r * HP + j] = (_Float16)(ov * tanhf(cn));
        }
        __syncthreads();

        // FC: out_tile[16x160] = h @ fcW^T  (waves 0..9, one 16-col tile each)
        if (wid < NT_FC) {
            v8f acc = {};
#pragma unroll
            for (int ks = 0; ks < KS_H; ++ks) {
                v16h a = load_a(sH, HP, ks, lane);
                v16h b = load_b(sFc, KS_H, wid, ks, lane);
                acc = __builtin_amdgcn_wmma_f32_16x16x32_f16(false, a, false, b,
                                                             (short)0, acc, false, false);
            }
            store_acc(sGate, GDIM, wid, lane, acc);   // cols 0..159 of gates buf
        }
        __syncthreads();

        // add bias, emit to d_out (N,C,T,V,M), and feed back as next f16 input
        for (int e = tid; e < 16 * FDIM; e += NTHREADS) {
            int r = e / FDIM, f = e % FDIM;
            float v = sGate[r * GDIM + f] + sFcb[f];
            out[(size_t)(n0 + r) * XB_N + (f / 50) * XB_C + t * 50 + (f % 50)] = v;
            sIn0[r * FP + f] = (_Float16)v;
        }
        __syncthreads();
    }
}

extern "C" void kernel_launch(void* const* d_in, const int* in_sizes, int n_in,
                              void* d_out, int out_size, void* d_ws, size_t ws_size,
                              hipStream_t stream) {
    (void)in_sizes; (void)n_in; (void)out_size; (void)d_ws; (void)ws_size;
    const float* x      = (const float*)d_in[0];
    const float* encWih = (const float*)d_in[1];
    const float* encWhh = (const float*)d_in[2];
    const float* encBih = (const float*)d_in[3];
    const float* encBhh = (const float*)d_in[4];
    const float* decWih = (const float*)d_in[5];
    const float* decWhh = (const float*)d_in[6];
    const float* decBih = (const float*)d_in[7];
    const float* decBhh = (const float*)d_in[8];
    const float* fcW    = (const float*)d_in[9];
    const float* fcB    = (const float*)d_in[10];

    dim3 grid(NBATCH / 16);     // 32 workgroups, one batch tile each
    dim3 block(NTHREADS);       // 25 waves of 32
    seq2seq_lstm_kernel<<<grid, block, SMEM_BYTES, stream>>>(
        x, encWih, encWhh, encBih, encBhh,
        decWih, decWhh, decBih, decBhh, fcW, fcB,
        (float*)d_out);
}
